// WaveNetKWS_1958505087232
// MI455X (gfx1250) — matrix operand order repaired and checked by reading, not compiled, not run on hardware
//
#include <hip/hip_runtime.h>

typedef __attribute__((ext_vector_type(16))) _Float16 v16h;
typedef __attribute__((ext_vector_type(8)))  float    v8f;

#define T0      8342           // 8192 + (1+2+4+8)*10
#define BATCH   32
#define RES_C   16
#define IN_C    40
#define SKIP_C  32
#define HID     128
#define NBLK    40
#define TPW     4              // t-tiles per wave in block kernel
#define XCAP    (RES_C * BATCH * T0)     // floats, x layout (b, t, 16)
#define SKIPSZ  (SKIP_C * BATCH * T0)    // floats, skip layout (b, t, 32)

__device__ __forceinline__ v8f wmma_f16(v16h a, v16h b, v8f c) {
  return __builtin_amdgcn_wmma_f32_16x16x32_f16(false, a, false, b, (short)0, c, false, false);
}
__device__ __forceinline__ float fast_sigmoid(float z) {
  return __builtin_amdgcn_rcpf(1.0f + __expf(-z));
}
__device__ __forceinline__ float fast_tanh(float y) {
  return __builtin_amdgcn_rcpf(1.0f + __expf(-2.0f * y)) * 2.0f - 1.0f;
}

// ---------------- init: zero stats + skip column t=0, seed out with dense_b ----------------
__global__ void init_kernel(float* __restrict__ skip, float* __restrict__ stats,
                            float* __restrict__ out, const float* __restrict__ dense_b) {
  int i = blockIdx.x * blockDim.x + threadIdx.x;
  if (i < NBLK * 32) stats[i] = 0.0f;
  if (i < BATCH * SKIP_C) {            // skip[b][t=0][ch] = 0 (never written by any block)
    int b = i >> 5, ch = i & 31;
    skip[(long long)b * T0 * 32 + ch] = 0.0f;
  }
  if (i < BATCH * 2) out[i] = dense_b[i & 1];
}

// ---------------- input 1x1 conv: 40 -> 16, output layout (b, t, 16) ----------------
__global__ void in_conv_kernel(const float* __restrict__ xin, const float* __restrict__ W,
                               const float* __restrict__ bias, float* __restrict__ x0) {
  __shared__ float Wl[IN_C * RES_C];
  __shared__ float bl[RES_C];
  for (int i = threadIdx.x; i < IN_C * RES_C; i += blockDim.x) Wl[i] = W[i];
  if (threadIdx.x < RES_C) bl[threadIdx.x] = bias[threadIdx.x];
  __syncthreads();
  int idx = blockIdx.x * blockDim.x + threadIdx.x;
  if (idx >= BATCH * T0) return;
  int b = idx / T0, t = idx - b * T0;
  float acc[RES_C];
#pragma unroll
  for (int o = 0; o < RES_C; o++) acc[o] = bl[o];
  const float* xp = xin + (long long)b * IN_C * T0 + t;
  for (int ci = 0; ci < IN_C; ci++) {
    float xv = xp[(long long)ci * T0];
#pragma unroll
    for (int o = 0; o < RES_C; o++) acc[o] += Wl[o * IN_C + ci] * xv;
  }
  float4* op = (float4*)(x0 + ((long long)b * T0 + t) * 16);
  op[0] = make_float4(acc[0], acc[1], acc[2], acc[3]);
  op[1] = make_float4(acc[4], acc[5], acc[6], acc[7]);
  op[2] = make_float4(acc[8], acc[9], acc[10], acc[11]);
  op[3] = make_float4(acc[12], acc[13], acc[14], acc[15]);
}

// ---------------- residual block ----------------
__global__ __launch_bounds__(128) void block_kernel(
    const float* __restrict__ xin, float* __restrict__ xout,
    float* __restrict__ skip, float* __restrict__ stats,
    const float* __restrict__ filt_W, const float* __restrict__ filt_b,
    const float* __restrict__ gate_W, const float* __restrict__ gate_b,
    const float* __restrict__ skip_W, const float* __restrict__ skip_b,
    const float* __restrict__ res_W,  const float* __restrict__ res_b,
    const float* __restrict__ bn_g,   const float* __restrict__ bn_b,
    int blk, int d, int offIn, int tilesT, int groupsT, float invCntPrev) {
  __shared__ float sc[16], sh[16];
  __shared__ float sstat[32];
  __shared__ float xn1[4][16][17];   // normalized tap-1 tile per wave (residual source)
  __shared__ float hld[4][16][17];   // gated activation tile per wave (transpose staging)

  const int tid = threadIdx.x;
  const int wv = tid >> 5, lane = tid & 31, hf = lane >> 4, col = lane & 15;

  if (tid < 32) sstat[tid] = 0.0f;
  if (tid < 16) {
    if (blk == 0) { sc[tid] = 1.0f; sh[tid] = 0.0f; }
    else {
      const float* ps = stats + (blk - 1) * 32;
      float mu  = ps[tid] * invCntPrev;
      float var = ps[16 + tid] * invCntPrev - mu * mu;
      float rs  = rsqrtf(var + 1e-5f);
      float g = bn_g[(blk - 1) * 16 + tid] * rs;
      sc[tid] = g;
      sh[tid] = bn_b[(blk - 1) * 16 + tid] - mu * g;
    }
  }
  __syncthreads();

  // hoisted A fragments (lane row M = col; element e -> K = 16*hf + e)
  const int M = col;
  v16h af, ag, as0, as1, ar;
#pragma unroll
  for (int e = 0; e < 16; e++) {
    int K = 16 * hf + e;
    int tap = K >> 4, ci = K & 15;
    af[e] = (_Float16)filt_W[((blk * 16 + M) * 16 + ci) * 2 + tap];
    ag[e] = (_Float16)gate_W[((blk * 16 + M) * 16 + ci) * 2 + tap];
    float w0 = 0.f, w1 = 0.f, wr = 0.f;
    if (K < 16) {
      w0 = skip_W[(blk * 32 + M) * 16 + K];
      w1 = skip_W[(blk * 32 + 16 + M) * 16 + K];
      wr = res_W[(blk * 16 + M) * 16 + K];
    }
    as0[e] = (_Float16)w0; as1[e] = (_Float16)w1; ar[e] = (_Float16)wr;
  }
  float fb[8], gb[8], sb0[8], sb1[8], rb[8];
#pragma unroll
  for (int j = 0; j < 8; j++) {
    int mo = j + 8 * hf;
    fb[j] = filt_b[blk * 16 + mo];  gb[j] = gate_b[blk * 16 + mo];
    sb0[j] = skip_b[blk * 32 + mo]; sb1[j] = skip_b[blk * 32 + 16 + mo];
    rb[j] = res_b[blk * 16 + mo];
  }

  int gid = blockIdx.x * 4 + wv;
  bool liveW = gid < BATCH * groupsT;
  int gidc = liveW ? gid : BATCH * groupsT - 1;
  int b = gidc / groupsT, tg = gidc - b * groupsT;

  float ps[8], ss[8];
#pragma unroll
  for (int j = 0; j < 8; j++) { ps[j] = 0.f; ss[j] = 0.f; }
  const v8f cz = {0.f, 0.f, 0.f, 0.f, 0.f, 0.f, 0.f, 0.f};

  for (int k = 0; k < TPW; k++) {
    int tt = tg * TPW + k;
    int t = offIn + d + tt * 16 + col;
    bool valid = liveW && (tt < tilesT) && (t < T0);
    int tc = (t < T0) ? t : (T0 - 1);
    int tl = tc - (hf ? 0 : d);   // lanes 0-15 tap0 (t-d), lanes 16-31 tap1 (t)

    // contiguous 64B per lane: 4x b128 loads, then BN on the fly
    const float4* xp = (const float4*)(xin + ((long long)b * T0 + tl) * 16);
    float4 q0 = xp[0], q1 = xp[1], q2 = xp[2], q3 = xp[3];
    float val[16] = {q0.x, q0.y, q0.z, q0.w, q1.x, q1.y, q1.z, q1.w,
                     q2.x, q2.y, q2.z, q2.w, q3.x, q3.y, q3.z, q3.w};
    v16h bx;
#pragma unroll
    for (int c = 0; c < 16; c++) {
      float v = val[c] * sc[c] + sh[c];
      val[c] = v;
      bx[c] = (_Float16)v;
    }
    if (hf) {
#pragma unroll
      for (int c = 0; c < 16; c++) xn1[wv][c][col] = val[c];
    }

    v8f accf = wmma_f16(af, bx, cz);
    v8f accg = wmma_f16(ag, bx, cz);

#pragma unroll
    for (int j = 0; j < 8; j++) {
      int mo = j + 8 * hf;
      float th = fast_tanh(accf[j] + fb[j]);
      float sg = fast_sigmoid(accg[j] + gb[j]);
      hld[wv][mo][col] = th * sg;
    }
    __syncthreads();

    v16h bh;   // K=16 zero-padded to 32: element e -> K = 16*hf + e
#pragma unroll
    for (int e = 0; e < 16; e++) bh[e] = (_Float16)(hf ? 0.0f : hld[wv][e][col]);

    v8f s0 = wmma_f16(as0, bh, cz);
    v8f s1 = wmma_f16(as1, bh, cz);
    v8f rr = wmma_f16(ar,  bh, cz);

    if (valid) {
      float so0[8], so1[8], xo[8];
#pragma unroll
      for (int j = 0; j < 8; j++) {
        so0[j] = s0[j] + sb0[j];
        so1[j] = s1[j] + sb1[j];
        float xv = xn1[wv][j + 8 * hf][col] + rr[j] + rb[j];
        xo[j] = xv;
        ps[j] += xv;
        ss[j] += xv * xv;
      }
      long long bt = (long long)b * T0 + tc;
      float4* sp0 = (float4*)(skip + bt * 32 + 8 * hf);       // channels 8hf..8hf+7
      float4* sp1 = (float4*)(skip + bt * 32 + 16 + 8 * hf);  // channels 16+8hf..
      float4 w0 = make_float4(so0[0], so0[1], so0[2], so0[3]);
      float4 w1 = make_float4(so0[4], so0[5], so0[6], so0[7]);
      float4 w2 = make_float4(so1[0], so1[1], so1[2], so1[3]);
      float4 w3 = make_float4(so1[4], so1[5], so1[6], so1[7]);
      if (blk == 0) {                 // first block: plain store, skip needs no zero-init
        sp0[0] = w0; sp0[1] = w1; sp1[0] = w2; sp1[1] = w3;
      } else {
        float4 a0 = sp0[0], a1 = sp0[1], a2 = sp1[0], a3 = sp1[1];
        sp0[0] = make_float4(a0.x + w0.x, a0.y + w0.y, a0.z + w0.z, a0.w + w0.w);
        sp0[1] = make_float4(a1.x + w1.x, a1.y + w1.y, a1.z + w1.z, a1.w + w1.w);
        sp1[0] = make_float4(a2.x + w2.x, a2.y + w2.y, a2.z + w2.z, a2.w + w2.w);
        sp1[1] = make_float4(a3.x + w3.x, a3.y + w3.y, a3.z + w3.z, a3.w + w3.w);
      }
      float4* xp4 = (float4*)(xout + bt * 16 + 8 * hf);
      xp4[0] = make_float4(xo[0], xo[1], xo[2], xo[3]);
      xp4[1] = make_float4(xo[4], xo[5], xo[6], xo[7]);
    }
    __syncthreads();
  }

  // BN stats: shuffle-reduce over the 16-lane halves, then LDS, then 32 global atomics / WG
#pragma unroll
  for (int j = 0; j < 8; j++) {
#pragma unroll
    for (int m = 1; m < 16; m <<= 1) {
      ps[j] += __shfl_xor(ps[j], m, 32);
      ss[j] += __shfl_xor(ss[j], m, 32);
    }
  }
  if (col == 0) {
#pragma unroll
    for (int j = 0; j < 8; j++) {
      atomicAdd(&sstat[j + 8 * hf], ps[j]);
      atomicAdd(&sstat[16 + j + 8 * hf], ss[j]);
    }
  }
  __syncthreads();
  if (tid < 32) atomicAdd(&stats[blk * 32 + tid], sstat[tid]);
}

// ---------------- output head: relu(skip) -> 128 -> 2 -> dense flatten ----------------
__global__ __launch_bounds__(128) void out_kernel(
    const float* __restrict__ skip,
    const float* __restrict__ out1_W, const float* __restrict__ out1_b,
    const float* __restrict__ out2_W, const float* __restrict__ out2_b,
    const float* __restrict__ dense_W, float* __restrict__ out) {
  __shared__ float h1s[4][HID][17];
  const int tid = threadIdx.x;
  const int wv = tid >> 5, lane = tid & 31, hf = lane >> 4, col = lane & 15;
  const int tilesT = (T0 + 15) / 16;
  const int numTiles = BATCH * tilesT;
  int tileRaw = blockIdx.x * 4 + wv;
  bool live = tileRaw < numTiles;
  int tile = live ? tileRaw : numTiles - 1;
  int b = tile / tilesT, tt = tile - b * tilesT;
  int t = tt * 16 + col;
  bool valid = live && (t < T0);
  int tc = (t < T0) ? t : (T0 - 1);

  // B fragment: relu(skip); lane reads its 16 channels contiguously (4x b128)
  const float4* sp = (const float4*)(skip + ((long long)b * T0 + tc) * 32 + 16 * hf);
  float4 r0 = sp[0], r1 = sp[1], r2 = sp[2], r3 = sp[3];
  float sv[16] = {r0.x, r0.y, r0.z, r0.w, r1.x, r1.y, r1.z, r1.w,
                  r2.x, r2.y, r2.z, r2.w, r3.x, r3.y, r3.z, r3.w};
  v16h bs;
#pragma unroll
  for (int e = 0; e < 16; e++) bs[e] = (_Float16)fmaxf(sv[e], 0.0f);

  const v8f cz = {0.f, 0.f, 0.f, 0.f, 0.f, 0.f, 0.f, 0.f};
  const int M = col;

  for (int m = 0; m < 8; m++) {        // 128 hidden rows = 8 M-tiles, K = 32 exact
    v16h am;
#pragma unroll
    for (int e = 0; e < 16; e++) {
      int K = 16 * hf + e;
      am[e] = (_Float16)out1_W[(m * 16 + M) * 32 + K];
    }
    v8f a = wmma_f16(am, bs, cz);
#pragma unroll
    for (int j = 0; j < 8; j++) {
      int row = m * 16 + j + 8 * hf;
      h1s[wv][row][col] = fmaxf(a[j] + out1_b[row], 0.0f);
    }
  }
  __syncthreads();

  v8f acc = cz;                         // 2x128 GEMM: 4 chained K=32 WMMAs, M padded to 16
#pragma unroll
  for (int q = 0; q < 4; q++) {
    v16h aq, bq;
#pragma unroll
    for (int e = 0; e < 16; e++) {
      int K = 16 * hf + e;
      aq[e] = (_Float16)((M < 2) ? out2_W[M * 128 + q * 32 + K] : 0.0f);
      bq[e] = (_Float16)h1s[wv][q * 32 + 16 * hf + e][col];
    }
    acc = wmma_f16(aq, bq, acc);
  }

  float p0 = 0.0f, p1 = 0.0f;           // dense flatten: dW[cls][c*T0 + t]
  if (hf == 0 && valid) {
    float c0 = acc[0] + out2_b[0];
    float c1 = acc[1] + out2_b[1];
    p0 = dense_W[t] * c0 + dense_W[T0 + t] * c1;
    p1 = dense_W[2 * T0 + t] * c0 + dense_W[3 * T0 + t] * c1;
  }
#pragma unroll
  for (int m = 16; m >= 1; m >>= 1) { p0 += __shfl_xor(p0, m, 32); p1 += __shfl_xor(p1, m, 32); }
  if (lane == 0 && live) {
    atomicAdd(&out[b * 2 + 0], p0);
    atomicAdd(&out[b * 2 + 1], p1);
  }
}

extern "C" void kernel_launch(void* const* d_in, const int* in_sizes, int n_in,
                              void* d_out, int out_size, void* d_ws, size_t ws_size,
                              hipStream_t stream) {
  const float* x       = (const float*)d_in[0];
  const float* in_W    = (const float*)d_in[1];
  const float* in_b    = (const float*)d_in[2];
  const float* filt_W  = (const float*)d_in[3];
  const float* filt_b  = (const float*)d_in[4];
  const float* gate_W  = (const float*)d_in[5];
  const float* gate_b  = (const float*)d_in[6];
  const float* skip_W  = (const float*)d_in[7];
  const float* skip_b  = (const float*)d_in[8];
  const float* res_W   = (const float*)d_in[9];
  const float* res_b   = (const float*)d_in[10];
  const float* bn_g    = (const float*)d_in[11];
  const float* bn_b    = (const float*)d_in[12];
  const float* out1_W  = (const float*)d_in[13];
  const float* out1_b  = (const float*)d_in[14];
  const float* out2_W  = (const float*)d_in[15];
  const float* out2_b  = (const float*)d_in[16];
  const float* dense_W = (const float*)d_in[17];
  const float* dense_b = (const float*)d_in[18];
  float* out = (float*)d_out;

  float* ws    = (float*)d_ws;
  float* x0    = ws;                 // ping, (b,t,16)
  float* x1    = x0 + XCAP;          // pong
  float* skip  = x1 + XCAP;          // (b,t,32)
  float* stats = skip + SKIPSZ;      // NBLK * 32 floats (sum[16], sumsq[16])

  init_kernel<<<8, 256, 0, stream>>>(skip, stats, out, dense_b);
  in_conv_kernel<<<(BATCH * T0 + 255) / 256, 256, 0, stream>>>(x, in_W, in_b, x0);

  int dil[NBLK], offs[NBLK + 1];
  offs[0] = 0;
  for (int i = 0; i < NBLK; i++) { dil[i] = 1 << (i & 3); offs[i + 1] = offs[i] + dil[i]; }

  float* bufs[2] = {x0, x1};
  for (int i = 0; i < NBLK; i++) {
    int Tout = T0 - offs[i + 1];
    int tilesT = (Tout + 15) / 16;
    int groupsT = (tilesT + TPW - 1) / TPW;
    int numGroups = BATCH * groupsT;
    float invCntPrev = (i == 0) ? 0.0f : 1.0f / (32.0f * (float)(T0 - offs[i]));
    block_kernel<<<(numGroups + 3) / 4, 128, 0, stream>>>(
        bufs[i & 1], bufs[(i + 1) & 1], skip, stats,
        filt_W, filt_b, gate_W, gate_b, skip_W, skip_b, res_W, res_b,
        bn_g, bn_b, i, dil[i], offs[i], tilesT, groupsT, invCntPrev);
  }
  {
    int tilesT = (T0 + 15) / 16;
    int numTiles = BATCH * tilesT;
    out_kernel<<<(numTiles + 3) / 4, 128, 0, stream>>>(
        skip, out1_W, out1_b, out2_W, out2_b, dense_W, out);
  }
}
